// GraphConv2_16518444221042
// MI455X (gfx1250) — matrix operand order repaired and checked by reading, not compiled, not run on hardware
//
#include <hip/hip_runtime.h>
#include <hip/hip_bf16.h>

typedef __attribute__((ext_vector_type(16))) __bf16       v16bf;
typedef __attribute__((ext_vector_type(8)))  float        v8f;
typedef __attribute__((ext_vector_type(4)))  unsigned int u32x4;

static __device__ inline v8f vzero8() {
    v8f z = {0.f,0.f,0.f,0.f,0.f,0.f,0.f,0.f};
    return z;
}

// ---------------------------------------------------------------------------
// CDNA5 WMMA bf16 fragment loaders (wave32), vectorized as 2x ds_load_b128.
//  A 16x32: lane m=lane&15, kb=(lane>>4)*8; values = row[kb..kb+8) ++ row[16+kb..16+kb+8)
//  B 32x16: lane n=lane&15, kb=(lane>>4)*8; values = col-major-K row[kb..kb+8) ++ row[16+kb..16+kb+8)
//  C/D 16x16 f32: element r -> (m = r + (lane>>4)*8, n = lane&15)
// Row strides must be multiples of 8 elements (16 B) for alignment.
// ---------------------------------------------------------------------------
static __device__ inline v16bf load_a_frag(const __bf16* As, int lda) {
    int lane = threadIdx.x & 31;
    const __bf16* row = As + (lane & 15) * lda;
    int kb = (lane >> 4) * 8;
    union { u32x4 q[2]; v16bf f; } u;
    u.q[0] = *(const u32x4*)(row + kb);
    u.q[1] = *(const u32x4*)(row + 16 + kb);
    return u.f;
}

// B stored K-contiguous per column: BT[n][k], ldb = row stride in elements.
static __device__ inline v16bf load_bT_frag(const __bf16* BT, int ldb) {
    int lane = threadIdx.x & 31;
    const __bf16* row = BT + (lane & 15) * ldb;
    int kb = (lane >> 4) * 8;
    union { u32x4 q[2]; v16bf f; } u;
    u.q[0] = *(const u32x4*)(row + kb);
    u.q[1] = *(const u32x4*)(row + 16 + kb);
    return u.f;
}

#define WMMA_BF16(a, b, c) \
    __builtin_amdgcn_wmma_f32_16x16x32_bf16(false, (a), false, (b), (short)0, (c), false, false)

// ---------------------------------------------------------------------------
__global__ void zero_kernel(float* __restrict__ p, int n) {
    int i = blockIdx.x * blockDim.x + threadIdx.x;
    if (i < n) p[i] = 0.f;
}

__global__ void convert_weights_kernel(const float* __restrict__ W1,
                                       const float* __restrict__ W2,
                                       const float* __restrict__ W3,
                                       const float* __restrict__ W4,
                                       __bf16* __restrict__ W1b,
                                       __bf16* __restrict__ W2b,
                                       __bf16* __restrict__ W3b,
                                       __bf16* __restrict__ W4b) {
    int i = blockIdx.x * blockDim.x + threadIdx.x;
    if (i < 65536)  W1b[i] = (__bf16)W1[i];
    if (i < 65536)  W2b[i] = (__bf16)W2[i];
    if (i < 32768)  W3b[i] = (__bf16)W3[i];
    if (i < 131072) W4b[i] = (__bf16)W4[i];
}

// ---------------------------------------------------------------------------
// Tiled WMMA GEMM: Y[E x NOUT] = act(A[E x 256]) @ Wb[256 x NOUT]
//   MODE 0: A row e = concat(node[src(e)], node[dst(e)])  (gather)
//   MODE 1: A row e = relu(aff_a * Yprev[e] + aff_c)      (BN-folded affine)
// Block 256 thr (8 waves), tile M=128, K stepped by 32, all N resident.
// B is staged K-contiguous (transposed) so fragments are 2x ds_load_b128.
// ---------------------------------------------------------------------------
template <int NOUT, int MODE>
__global__ __launch_bounds__(256) void gemm_kernel(
    const float*      __restrict__ Ain,
    const long long*  __restrict__ eidx,
    const float*      __restrict__ node,
    const float*      __restrict__ aff_a,
    const float*      __restrict__ aff_c,
    const __bf16*     __restrict__ Wb,     // [256 x NOUT] row-major
    float*            __restrict__ Y,      // [E x NOUT]
    int E) {
    __shared__ __bf16 As[128][40];      // 128 rows x 32 K (padded to 40)
    __shared__ __bf16 BsT[NOUT][40];    // NOUT cols x 32 K (K-contiguous)

    const int tid  = threadIdx.x;
    const int wave = tid >> 5;
    const int lane = tid & 31;
    const size_t base = (size_t)blockIdx.x * 128;

    constexpr int NT = NOUT / 16;
    v8f acc[NT];
#pragma unroll
    for (int i = 0; i < NT; ++i) acc[i] = vzero8();

    for (int kt = 0; kt < 8; ++kt) {     // K = 256
        // stage A tile (bf16, zero-pad tail rows)
        for (int i = tid; i < 128 * 32; i += 256) {
            int r = i >> 5, c = i & 31;
            size_t e = base + r;
            int k = kt * 32 + c;
            float v = 0.f;
            if (e < (size_t)E) {
                if (MODE == 0) {
                    long long nidx = eidx[e * 2 + (k < 128 ? 0 : 1)];
                    v = node[(size_t)nidx * 128 + (k & 127)];
                } else {
                    float y = Ain[e * 256 + k];
                    v = fmaxf(0.f, aff_a[k] * y + aff_c[k]);
                }
            }
            As[r][c] = (__bf16)v;
        }
        // stage B tile transposed: BsT[n][k] = W[kt*32+k][n]
        for (int i = tid; i < 32 * NOUT; i += 256) {
            int n = i % NOUT, k = i / NOUT;     // consecutive tid -> consecutive n (coalesced)
            BsT[n][k] = Wb[(size_t)(kt * 32 + k) * NOUT + n];
        }
        __syncthreads();

        v16bf af = load_a_frag(&As[wave * 16][0], 40);
#pragma unroll
        for (int nt = 0; nt < NT; ++nt) {
            v16bf bf = load_bT_frag(&BsT[nt * 16][0], 40);
            acc[nt] = WMMA_BF16(af, bf, acc[nt]);
        }
        __syncthreads();
    }

#pragma unroll
    for (int nt = 0; nt < NT; ++nt) {
#pragma unroll
        for (int r = 0; r < 8; ++r) {
            size_t row = base + wave * 16 + ((lane >> 4) * 8) + r;
            if (row < (size_t)E)
                Y[row * NOUT + nt * 16 + (lane & 15)] = acc[nt][r];
        }
    }
}

// ---------------------------------------------------------------------------
// Column sums / sums-of-squares for BN stats. blockDim.x == C.
// ---------------------------------------------------------------------------
__global__ void col_stats_kernel(const float* __restrict__ Y, int E, int C,
                                 int rowsPerBlock,
                                 float* __restrict__ sum, float* __restrict__ sumsq) {
    int c = threadIdx.x;
    int r0 = blockIdx.x * rowsPerBlock;
    int r1 = min(E, r0 + rowsPerBlock);
    float s = 0.f, s2 = 0.f;
    for (int r = r0; r < r1; ++r) {
        float v = Y[(size_t)r * C + c];
        s += v; s2 += v * v;
    }
    atomicAdd(&sum[c], s);
    atomicAdd(&sumsq[c], s2);
}

__global__ void finalize_affine_kernel(const float* __restrict__ sum,
                                       const float* __restrict__ sumsq,
                                       const float* __restrict__ g,
                                       const float* __restrict__ b,
                                       int C, float invE,
                                       float* __restrict__ a, float* __restrict__ c) {
    int j = blockIdx.x * blockDim.x + threadIdx.x;
    if (j >= C) return;
    float mu  = sum[j] * invE;
    float var = fmaxf(0.f, sumsq[j] * invE - mu * mu);
    float s   = g[j] * rsqrtf(var + 1e-5f);
    a[j] = s;
    c[j] = b[j] - s * mu;
}

// ---------------------------------------------------------------------------
// Gram kernel: G[128x128] += H3^T H3, S[128] += colsum(H3),
// H3 = relu(a3*Y3+c3). H3 chunk staged edge-contiguous (transposed) once so
// BOTH A and B fragments are contiguous 2x ds_load_b128.
// ---------------------------------------------------------------------------
__global__ __launch_bounds__(256) void gram_kernel(
    const float* __restrict__ Y3, const float* __restrict__ a3,
    const float* __restrict__ c3, int E, int edgesPerBlock,
    float* __restrict__ G, float* __restrict__ S) {
    __shared__ __bf16 HsT[128][40];   // [dim][edge-in-chunk], 32 edges
    const int tid  = threadIdx.x;
    const int wave = tid >> 5;
    const int lane = tid & 31;

    v8f acc[8];
#pragma unroll
    for (int i = 0; i < 8; ++i) acc[i] = vzero8();
    float colsum = 0.f;

    int e0 = blockIdx.x * edgesPerBlock;
    int e1 = min(E, e0 + edgesPerBlock);
    for (int eb = e0; eb < e1; eb += 32) {
        for (int i = tid; i < 32 * 128; i += 256) {
            int d = i & 127, r = i >> 7;   // consecutive tid -> consecutive d (coalesced)
            int e = eb + r;
            float v = (e < E) ? fmaxf(0.f, a3[d] * Y3[(size_t)e * 128 + d] + c3[d]) : 0.f;
            HsT[d][r] = (__bf16)v;
        }
        __syncthreads();

        // A[m][k] = H3[k][wave*16+m] = HsT[wave*16+m][k]  (contiguous in k)
        v16bf af = load_a_frag(&HsT[wave * 16][0], 40);
#pragma unroll
        for (int tc = 0; tc < 8; ++tc) {
            // B[k][n] = H3[k][tc*16+n] = HsT[tc*16+n][k]  (contiguous in k)
            v16bf bf = load_bT_frag(&HsT[tc * 16][0], 40);
            acc[tc] = WMMA_BF16(af, bf, acc[tc]);
        }
        if (tid < 128) {
            float cs = 0.f;
#pragma unroll
            for (int r = 0; r < 32; ++r) cs += (float)HsT[tid][r];
            colsum += cs;
        }
        __syncthreads();
    }

#pragma unroll
    for (int tc = 0; tc < 8; ++tc)
#pragma unroll
        for (int r = 0; r < 8; ++r) {
            int m = wave * 16 + r + ((lane >> 4) * 8);
            int n = tc * 16 + (lane & 15);
            atomicAdd(&G[m * 128 + n], acc[tc][r]);
        }
    if (tid < 128) atomicAdd(&S[tid], colsum);
}

// ---------------------------------------------------------------------------
// Layer-4 BN affine from Gram: mu_j=(S/E).w_j, var_j = w_j^T(G/E)w_j - mu^2.
// ---------------------------------------------------------------------------
__global__ void finalize_l4_kernel(const float* __restrict__ G,
                                   const float* __restrict__ S,
                                   const float* __restrict__ W4,
                                   const float* __restrict__ g4,
                                   const float* __restrict__ b4,
                                   float invE,
                                   float* __restrict__ a4, float* __restrict__ c4) {
    int j = blockIdx.x * blockDim.x + threadIdx.x;
    if (j >= 1024) return;
    float w[128];
    for (int k = 0; k < 128; ++k) w[k] = W4[(size_t)k * 1024 + j];
    float mu = 0.f;
    for (int k = 0; k < 128; ++k) mu += S[k] * w[k];
    mu *= invE;
    float m2 = 0.f;
    for (int k = 0; k < 128; ++k) {
        float gk = 0.f;
        for (int l = 0; l < 128; ++l) gk += G[k * 128 + l] * w[l];
        m2 += w[k] * gk;
    }
    m2 *= invE;
    float var = fmaxf(0.f, m2 - mu * mu);
    float s = g4[j] * rsqrtf(var + 1e-5f);
    a4[j] = s;
    c4[j] = b4[j] - s * mu;
}

// ---------------------------------------------------------------------------
// Fused final kernel: per 128-edge block
//   h3 -> bf16 LDS; per d: y4 column block via WMMA (K=128, N=32),
//   msg += edge[:,d] * (a4*y4 + c4); msg = relu(msg + bias); GRU; 2x store.
// W4 column chunk staged K-contiguous so B frags are 2x ds_load_b128.
// ---------------------------------------------------------------------------
__global__ __launch_bounds__(256) void final_fused_kernel(
    const float* __restrict__ Y3, const float* __restrict__ a3,
    const float* __restrict__ c3, const __bf16* __restrict__ W4b,
    const float* __restrict__ a4, const float* __restrict__ c4,
    const float* __restrict__ edge, const float* __restrict__ bias,
    const float* __restrict__ hidden,
    const float* __restrict__ W_ih, const float* __restrict__ W_hh,
    const float* __restrict__ b_ih, const float* __restrict__ b_hh,
    float* __restrict__ out, int E) {
    __shared__ __bf16 Hs[128][136];   // h3 tile [edge][dim]      ~34.8 KB
    __shared__ float  Es[128][33];    // edge tile, reused as msg ~16.9 KB
    __shared__ __bf16 WsT[32][136];   // W4 chunk [col][k]        ~8.7 KB

    const int tid  = threadIdx.x;
    const int wave = tid >> 5;
    const int lane = tid & 31;
    const size_t base = (size_t)blockIdx.x * 128;

    for (int i = tid; i < 128 * 128; i += 256) {
        int r = i >> 7, d = i & 127;
        size_t e = base + r;
        float v = (e < (size_t)E) ? fmaxf(0.f, a3[d] * Y3[e * 128 + d] + c3[d]) : 0.f;
        Hs[r][d] = (__bf16)v;
    }
    for (int i = tid; i < 128 * 32; i += 256) {
        int r = i >> 5, c = i & 31;
        size_t e = base + r;
        Es[r][c] = (e < (size_t)E) ? edge[e * 32 + c] : 0.f;
    }
    __syncthreads();

    // Preload this wave's 4 A fragments (16 edges x 128 dims), reused for all d.
    v16bf afr[4];
#pragma unroll
    for (int kt = 0; kt < 4; ++kt)
        afr[kt] = load_a_frag(&Hs[wave * 16][kt * 32], 136);

    v8f macc[2];
    macc[0] = vzero8(); macc[1] = vzero8();
    const int mrow0 = wave * 16 + ((lane >> 4) * 8);
    const int ncol  = lane & 15;

    for (int d = 0; d < 32; ++d) {
        // stage W4 chunk transposed: WsT[c][k] = W4[k][d*32+c]
        for (int i = tid; i < 128 * 32; i += 256) {
            int c = i & 31, k = i >> 5;    // consecutive tid -> consecutive c (coalesced)
            WsT[c][k] = W4b[(size_t)k * 1024 + d * 32 + c];
        }
        __syncthreads();

        v8f yt[2];
        yt[0] = vzero8(); yt[1] = vzero8();
#pragma unroll
        for (int kt = 0; kt < 4; ++kt) {
#pragma unroll
            for (int nt = 0; nt < 2; ++nt) {
                v16bf bf = load_bT_frag(&WsT[nt * 16][kt * 32], 136);
                yt[nt] = WMMA_BF16(afr[kt], bf, yt[nt]);
            }
        }
        // fold BN affine + contract with edge[:,d]
#pragma unroll
        for (int nt = 0; nt < 2; ++nt) {
            int j = d * 32 + nt * 16 + ncol;
            float aj = a4[j], cj = c4[j];
#pragma unroll
            for (int r = 0; r < 8; ++r) {
                float ev = Es[mrow0 + r][d];
                macc[nt][r] += ev * (aj * yt[nt][r] + cj);
            }
        }
        __syncthreads();
    }

    // msg = relu(macc + bias) -> LDS (reuse edge buffer)
#pragma unroll
    for (int nt = 0; nt < 2; ++nt) {
        int f = nt * 16 + ncol;
        float bf_ = bias[f];
#pragma unroll
        for (int r = 0; r < 8; ++r)
            Es[mrow0 + r][f] = fmaxf(0.f, macc[nt][r] + bf_);
    }
    __syncthreads();

    // GRU step: one (edge,col) element per thread-iteration.
    for (int it = 0; it < 16; ++it) {
        int idx = it * 256 + tid;
        int el = idx >> 5, f = idx & 31;
        size_t e = base + el;
        if (e < (size_t)E) {
            float gi_r = b_ih[f], gi_z = b_ih[32 + f], gi_n = b_ih[64 + f];
            float gh_r = b_hh[f], gh_z = b_hh[32 + f], gh_n = b_hh[64 + f];
#pragma unroll 8
            for (int k = 0; k < 32; ++k) {
                float mk = Es[el][k];
                float hk = hidden[e * 32 + k];
                gi_r += mk * W_ih[f * 32 + k];
                gi_z += mk * W_ih[(32 + f) * 32 + k];
                gi_n += mk * W_ih[(64 + f) * 32 + k];
                gh_r += hk * W_hh[f * 32 + k];
                gh_z += hk * W_hh[(32 + f) * 32 + k];
                gh_n += hk * W_hh[(64 + f) * 32 + k];
            }
            float rr = 1.f / (1.f + __expf(-(gi_r + gh_r)));
            float zz = 1.f / (1.f + __expf(-(gi_z + gh_z)));
            float nn = tanhf(gi_n + rr * gh_n);
            float h0v = hidden[e * 32 + f];
            float hn = (1.f - zz) * nn + zz * h0v;
            out[e * 32 + f] = hn;                      // output 0: h_new
            out[(size_t)E * 32 + e * 32 + f] = hn;     // output 1: h_new[None]
        }
    }
}

// ---------------------------------------------------------------------------
extern "C" void kernel_launch(void* const* d_in, const int* in_sizes, int n_in,
                              void* d_out, int out_size, void* d_ws, size_t ws_size,
                              hipStream_t stream) {
    const float*     node   = (const float*)d_in[0];
    const long long* eidx   = (const long long*)d_in[1];
    const float*     edge   = (const float*)d_in[2];
    const float*     hidden = (const float*)d_in[3];
    const float* W1 = (const float*)d_in[4];
    const float* g1 = (const float*)d_in[5];
    const float* b1 = (const float*)d_in[6];
    const float* W2 = (const float*)d_in[7];
    const float* g2 = (const float*)d_in[8];
    const float* b2 = (const float*)d_in[9];
    const float* W3 = (const float*)d_in[10];
    const float* g3 = (const float*)d_in[11];
    const float* b3 = (const float*)d_in[12];
    const float* W4 = (const float*)d_in[13];
    const float* g4 = (const float*)d_in[14];
    const float* b4 = (const float*)d_in[15];
    const float* bias = (const float*)d_in[16];
    const float* W_ih = (const float*)d_in[17];
    const float* W_hh = (const float*)d_in[18];
    const float* b_ih = (const float*)d_in[19];
    const float* b_hh = (const float*)d_in[20];
    float* out = (float*)d_out;

    const int E = in_sizes[2] / 32;
    const float invE = 1.f / (float)E;

    // ---- workspace layout ----
    char* p = (char*)d_ws;
    size_t off = 0;
    auto alloc = [&](size_t bytes) -> char* {
        char* q = p + off;
        off += (bytes + 255) & ~(size_t)255;
        return q;
    };
    float*  Y1  = (float*)alloc((size_t)E * 256 * sizeof(float));
    float*  Y2  = (float*)alloc((size_t)E * 256 * sizeof(float));
    float*  Y3  = (float*)alloc((size_t)E * 128 * sizeof(float));
    __bf16* W1b = (__bf16*)alloc(65536 * 2);
    __bf16* W2b = (__bf16*)alloc(65536 * 2);
    __bf16* W3b = (__bf16*)alloc(32768 * 2);
    __bf16* W4b = (__bf16*)alloc(131072 * 2);
    float*  st  = (float*)alloc(21120 * sizeof(float));
    float *sum1 = st,        *ss1 = st + 256,  *a1 = st + 512,  *c1 = st + 768;
    float *sum2 = st + 1024, *ss2 = st + 1280, *a2 = st + 1536, *c2 = st + 1792;
    float *sum3 = st + 2048, *ss3 = st + 2176, *a3 = st + 2304, *c3 = st + 2432;
    float *G = st + 2560, *S = st + 18944, *a4 = st + 19072, *c4 = st + 20096;

    const int tilesM = (E + 127) / 128;
    const int statsB = (E + 511) / 512;
    const int gramB  = (E + 2047) / 2048;

    zero_kernel<<<(21120 + 255) / 256, 256, 0, stream>>>(st, 21120);
    convert_weights_kernel<<<512, 256, 0, stream>>>(W1, W2, W3, W4, W1b, W2b, W3b, W4b);

    // Layer 1: gather + GEMM
    gemm_kernel<256, 0><<<tilesM, 256, 0, stream>>>(nullptr, eidx, node, nullptr, nullptr,
                                                    W1b, Y1, E);
    col_stats_kernel<<<statsB, 256, 0, stream>>>(Y1, E, 256, 512, sum1, ss1);
    finalize_affine_kernel<<<1, 256, 0, stream>>>(sum1, ss1, g1, b1, 256, invE, a1, c1);

    // Layer 2
    gemm_kernel<256, 1><<<tilesM, 256, 0, stream>>>(Y1, nullptr, nullptr, a1, c1,
                                                    W2b, Y2, E);
    col_stats_kernel<<<statsB, 256, 0, stream>>>(Y2, E, 256, 512, sum2, ss2);
    finalize_affine_kernel<<<1, 256, 0, stream>>>(sum2, ss2, g2, b2, 256, invE, a2, c2);

    // Layer 3
    gemm_kernel<128, 1><<<tilesM, 256, 0, stream>>>(Y2, nullptr, nullptr, a2, c2,
                                                    W3b, Y3, E);
    col_stats_kernel<<<statsB, 128, 0, stream>>>(Y3, E, 128, 512, sum3, ss3);
    finalize_affine_kernel<<<1, 128, 0, stream>>>(sum3, ss3, g3, b3, 128, invE, a3, c3);

    // Layer 4 stats via Gram of H3 (no Y4 materialization)
    gram_kernel<<<gramB, 256, 0, stream>>>(Y3, a3, c3, E, 2048, G, S);
    finalize_l4_kernel<<<4, 256, 0, stream>>>(G, S, W4, g4, b4, invE, a4, c4);

    // Fused layer4 + per-edge matmul + bias/relu + GRU
    final_fused_kernel<<<tilesM, 256, 0, stream>>>(Y3, a3, c3, W4b, a4, c4, edge, bias,
                                                   hidden, W_ih, W_hh, b_ih, b_hh, out, E);
    (void)n_in; (void)out_size; (void)ws_size;
}
